// AdvNet_73065983640345
// MI455X (gfx1250) — matrix operand order repaired and checked by reading, not compiled, not run on hardware
//
#include <hip/hip_runtime.h>
#include <hip/hip_bf16.h>

// ---------------------------------------------------------------------------
// Problem constants (from the reference)
// ---------------------------------------------------------------------------
#define B_ROWS    2048            // batch
#define N_TOT     150000          // N_USERS + N_ITEMS
#define EMB       64
#define NT        (N_TOT / 16)    // 9375 column tiles (150000 % 16 == 0)
#define NCH       32              // column chunks (gridDim.y)
#define NWAVES    8               // waves per block (256 threads, wave32)

typedef __attribute__((ext_vector_type(16))) __bf16 bf16x16;
typedef __attribute__((ext_vector_type(8)))  __bf16 bf16x8;
typedef __attribute__((ext_vector_type(4)))  __bf16 bf16x4;
typedef __attribute__((ext_vector_type(8))) float  f32x8;
typedef __attribute__((ext_vector_type(4))) float  f32x4;

#define NEG_INF    (-__builtin_inff())
#define M_SENTINEL (-1.0e30f)     // finite "minus infinity" for branchless LSE
#define LOG2E      1.44269504f

// raw hardware transcendentals (no OCML special-case guard code)
#define EXP2(x) __builtin_amdgcn_exp2f(x)
#define LOG2(x) __builtin_amdgcn_logf(x)

__device__ __forceinline__ unsigned hash32(unsigned x) {
    x ^= x >> 16; x *= 0x7feb352du;
    x ^= x >> 15; x *= 0x846ca68bu;
    x ^= x >> 16;
    return x;
}

// ---------------------------------------------------------------------------
// Kernel 1: fp32 -> bf16 conversion of the whole embedding table (L2-resident)
// ---------------------------------------------------------------------------
__global__ void cvt_embed_kernel(const float* __restrict__ src,
                                 __bf16* __restrict__ dst, int n4) {
    int i = blockIdx.x * blockDim.x + threadIdx.x;
    if (i < n4) {
        f32x4 v = ((const f32x4*)src)[i];
        bf16x4 o;
        o[0] = (__bf16)v.x; o[1] = (__bf16)v.y;
        o[2] = (__bf16)v.z; o[3] = (__bf16)v.w;
        ((bf16x4*)dst)[i] = o;
    }
}

// ---------------------------------------------------------------------------
// Kernel 2: q[b,k] = all_embed[u_id[b],k] + all_embed[pos_i_id[b],k]  (bf16)
// ---------------------------------------------------------------------------
__global__ void build_q_kernel(const int* __restrict__ u_id,
                               const int* __restrict__ p_id,
                               const float* __restrict__ A,
                               __bf16* __restrict__ q) {
    int t = blockIdx.x * blockDim.x + threadIdx.x;   // < B_ROWS*EMB
    int b = t >> 6;
    int k = t & 63;
    float v = A[(size_t)u_id[b] * EMB + k] + A[(size_t)p_id[b] * EMB + k];
    q[t] = (__bf16)v;
}

// ---------------------------------------------------------------------------
// Kernel 3: fused GEMM (bf16 WMMA) + online softmax + Gumbel-argmax partials.
//   grid = (B_ROWS/16, NCH), block = 256 (8 waves); each wave streams 16x16
//   score tiles via v_wmma_f32_16x16x32_bf16 (K=64 -> 2 WMMAs per tile).
//   All softmax/Gumbel state is kept in the base-2 (log2) domain so the
//   epilogue uses bare v_exp_f32 / v_log_f32 and stays fully branchless.
// ---------------------------------------------------------------------------
__global__ __launch_bounds__(256) void score_pass_kernel(
    const __bf16* __restrict__ E,   // [N_TOT][EMB] bf16
    const __bf16* __restrict__ Q,   // [B_ROWS][EMB] bf16
    float* __restrict__ Pm, float* __restrict__ Ps,
    float* __restrict__ Pg, float* __restrict__ Pl,
    int*   __restrict__ Pn)
{
    __shared__ float lds[NWAVES][16][5];

    const int tid  = threadIdx.x;
    // scalarize wave id so the tile loop runs on SALU
    const int wave = __builtin_amdgcn_readfirstlane(tid) >> 5;
    const int lane = tid & 31;
    const int li   = lane & 15;     // row (A) / column (B) within tile
    const int h    = lane >> 4;     // lane-half selector
    const int m0   = blockIdx.x * 16;
    const int ch   = blockIdx.y;

    // --- A operand (16 q-rows), ISA 16-bit A layout:
    //     element e -> K = (e>>3)*16 + h*8 + (e&7)
    union { bf16x16 v; bf16x8 half[2]; } qa_lo, qa_hi;
    {
        const __bf16* qrow = Q + (size_t)(m0 + li) * EMB;
        qa_lo.half[0] = *(const bf16x8*)(qrow +      h * 8);
        qa_lo.half[1] = *(const bf16x8*)(qrow + 16 + h * 8);
        qa_hi.half[0] = *(const bf16x8*)(qrow + 32 + h * 8);
        qa_hi.half[1] = *(const bf16x8*)(qrow + 48 + h * 8);
    }

    // per-lane running stats (base-2 domain) for the 8 rows of C this lane holds
    float m[8], s[8], bg[8], bl[8];
    int   bn[8];
    unsigned rh[8];                 // loop-invariant per-row hash
#pragma unroll
    for (int r = 0; r < 8; ++r) {
        m[r]  = M_SENTINEL; s[r] = 0.0f;
        bg[r] = NEG_INF; bl[r] = NEG_INF; bn[r] = 0x7fffffff;
        rh[r] = hash32((unsigned)(m0 + 8 * h + r) * 0x9E3779B9u);
    }

    const int g      = ch * NWAVES + wave;
    const int stride = NCH * NWAVES;

#pragma unroll 2
    for (int t = g; t < NT; t += stride) {
        const int ncol = t * 16 + li;
        const __bf16* erow = E + (size_t)ncol * EMB;

        // --- B operand (16 item columns = rows of E), ISA 16-bit B layout:
        //     element e -> K = (e>>3)*16 + h*8 + (e&7)
        union { bf16x16 v; bf16x8 half[2]; } bb_lo, bb_hi;
        bb_lo.half[0] = *(const bf16x8*)(erow +      h * 8);
        bb_lo.half[1] = *(const bf16x8*)(erow + 16 + h * 8);
        bb_hi.half[0] = *(const bf16x8*)(erow + 32 + h * 8);
        bb_hi.half[1] = *(const bf16x8*)(erow + 48 + h * 8);

        f32x8 c = {0.f, 0.f, 0.f, 0.f, 0.f, 0.f, 0.f, 0.f};
        c = __builtin_amdgcn_wmma_f32_16x16x32_bf16(false, qa_lo.v, false, bb_lo.v,
                                                    (short)0, c, false, false);
        c = __builtin_amdgcn_wmma_f32_16x16x32_bf16(false, qa_hi.v, false, bb_hi.v,
                                                    (short)0, c, false, false);

        const unsigned hn = hash32((unsigned)ncol + 0x85EBCA6Bu);

#pragma unroll
        for (int r = 0; r < 8; ++r) {
            float sc = c[r];
            // scores==0 -> -inf mask, then scale into log2 domain (1 cndmask+1 mul)
            float ml = ((sc == 0.0f) ? NEG_INF : sc) * LOG2E;

            // branchless online sum-of-exp2 (m finite sentinel => no NaN)
            float mn = fmaxf(m[r], ml);
            s[r] = s[r] * EXP2(m[r] - mn) + EXP2(ml - mn);
            m[r] = mn;

            // Gumbel-max in base-2: gs = ml - log2(-log2(u)); the uniform
            // additive constant log2(ln2) cancels in the argmax.
            unsigned hh = (rh[r] ^ hn) * 0x9E3779B9u;
            hh ^= hh >> 16;
            float u  = ((float)(hh >> 8) + 0.5f) * (1.0f / 16777216.0f);
            float gs = ml - LOG2(-LOG2(u));       // -inf stays -inf

            bool better = (gs > bg[r]) | ((gs == bg[r]) & (ncol < bn[r]));
            bg[r] = better ? gs   : bg[r];
            bl[r] = better ? ml   : bl[r];
            bn[r] = better ? ncol : bn[r];
        }
    }

    // --- reduce across the 16 lanes that share each row (xor 1,2,4,8)
#pragma unroll
    for (int off = 1; off < 16; off <<= 1) {
#pragma unroll
        for (int r = 0; r < 8; ++r) {
            float om = __shfl_xor(m[r],  off, 32);
            float os = __shfl_xor(s[r],  off, 32);
            float og = __shfl_xor(bg[r], off, 32);
            float ol = __shfl_xor(bl[r], off, 32);
            int   on = __shfl_xor(bn[r], off, 32);
            float M  = fmaxf(m[r], om);
            s[r] = s[r] * EXP2(m[r] - M) + os * EXP2(om - M);
            m[r] = M;
            bool better = (og > bg[r]) | ((og == bg[r]) & (on < bn[r]));
            bg[r] = better ? og : bg[r];
            bl[r] = better ? ol : bl[r];
            bn[r] = better ? on : bn[r];
        }
    }

    if (li == 0) {
#pragma unroll
        for (int r = 0; r < 8; ++r) {
            int row = 8 * h + r;
            lds[wave][row][0] = m[r];
            lds[wave][row][1] = s[r];
            lds[wave][row][2] = bg[r];
            lds[wave][row][3] = bl[r];
            lds[wave][row][4] = __int_as_float(bn[r]);
        }
    }
    __syncthreads();

    // --- reduce across the 8 waves; one partial record per (chunk, row)
    if (tid < 16) {
        float M = M_SENTINEL, S = 0.f, G = NEG_INF, L = NEG_INF;
        int   Nn = 0x7fffffff;
        for (int w = 0; w < NWAVES; ++w) {
            float om = lds[w][tid][0], os = lds[w][tid][1];
            float nM = fmaxf(M, om);
            S = S * EXP2(M - nM) + os * EXP2(om - nM);
            M = nM;
            float og = lds[w][tid][2];
            float ol = lds[w][tid][3];
            int   on = __float_as_int(lds[w][tid][4]);
            bool better = (og > G) | ((og == G) & (on < Nn));
            G  = better ? og : G;
            L  = better ? ol : L;
            Nn = better ? on : Nn;
        }
        size_t pi = (size_t)ch * B_ROWS + (m0 + tid);
        Pm[pi] = M; Ps[pi] = S; Pg[pi] = G; Pl[pi] = L; Pn[pi] = Nn;
    }
}

// ---------------------------------------------------------------------------
// Kernel 4: merge the NCH chunk partials per row, emit sample + probability
// ---------------------------------------------------------------------------
__global__ void finalize_kernel(const float* __restrict__ Pm,
                                const float* __restrict__ Ps,
                                const float* __restrict__ Pg,
                                const float* __restrict__ Pl,
                                const int*   __restrict__ Pn,
                                float* __restrict__ out) {
    int b = blockIdx.x * blockDim.x + threadIdx.x;
    if (b >= B_ROWS) return;
    float M = M_SENTINEL, S = 0.f, G = NEG_INF, L = NEG_INF;
    int   Nn = 0x7fffffff;
    for (int ch = 0; ch < NCH; ++ch) {
        size_t pi = (size_t)ch * B_ROWS + b;
        float om = Pm[pi], os = Ps[pi];
        float nM = fmaxf(M, om);
        S = S * EXP2(M - nM) + os * EXP2(om - nM);
        M = nM;
        float og = Pg[pi];
        int   on = Pn[pi];
        bool better = (og > G) | ((og == G) & (on < Nn));
        if (better) { G = og; L = Pl[pi]; Nn = on; }
    }
    // probability in base-2 domain: exp2(L - M) / S  (== exp(l - m)/s)
    float prob = (S > 0.f && L != NEG_INF) ? EXP2(L - M) / S : 0.0f;
    out[b]          = (float)Nn;   // samples
    out[B_ROWS + b] = prob;        // selected_neg_prob
}

// ---------------------------------------------------------------------------
// Entry point
// ---------------------------------------------------------------------------
extern "C" void kernel_launch(void* const* d_in, const int* in_sizes, int n_in,
                              void* d_out, int out_size, void* d_ws, size_t ws_size,
                              hipStream_t stream) {
    (void)in_sizes; (void)n_in; (void)out_size; (void)ws_size;

    const int*   u_id = (const int*)d_in[0];
    const int*   p_id = (const int*)d_in[1];
    // d_in[2] = train_mask (unused by the reference's output)
    const float* A    = (const float*)d_in[3];

    char* ws = (char*)d_ws;
    __bf16* E = (__bf16*)ws;                                   // 19.2 MB
    ws += (size_t)N_TOT * EMB * sizeof(__bf16);
    __bf16* Q = (__bf16*)ws;                                   // 256 KB
    ws += (size_t)B_ROWS * EMB * sizeof(__bf16);
    const size_t npart = (size_t)NCH * B_ROWS;
    float* Pm = (float*)ws; ws += npart * sizeof(float);
    float* Ps = (float*)ws; ws += npart * sizeof(float);
    float* Pg = (float*)ws; ws += npart * sizeof(float);
    float* Pl = (float*)ws; ws += npart * sizeof(float);
    int*   Pn = (int*)ws;

    int n4 = N_TOT * EMB / 4;
    cvt_embed_kernel<<<(n4 + 255) / 256, 256, 0, stream>>>(A, E, n4);
    build_q_kernel<<<(B_ROWS * EMB + 255) / 256, 256, 0, stream>>>(u_id, p_id, A, Q);

    dim3 grid(B_ROWS / 16, NCH);
    score_pass_kernel<<<grid, 256, 0, stream>>>(E, Q, Pm, Ps, Pg, Pl, Pn);

    finalize_kernel<<<(B_ROWS + 255) / 256, 256, 0, stream>>>(
        Pm, Ps, Pg, Pl, Pn, (float*)d_out);
}
